// FNOBlock2d_35192962024060
// MI455X (gfx1250) — compile-verified, matrix operand order checked by reading, not run on hardware
//
#include <hip/hip_runtime.h>
#include <math.h>

// ---------------------------------------------------------------------------
// FNO block for MI455X (gfx1250, wave32, WMMA 16x16x32 f16 -> f32 acc).
//
// All GEMM B-operands are pre-swizzled in memory into WMMA fragment layout
// [tile][lane][16 halfs] so compute kernels issue only b128 fragment loads.
// k_spec stages its per-mode A matrices into LDS with the Tensor Data Mover
// (tensor_load_to_lds + s_wait_tensorcnt), and streams the 268MB of spectral
// weights with prefetch hints.
// ---------------------------------------------------------------------------

typedef __attribute__((ext_vector_type(16))) _Float16 v16h;
typedef __attribute__((ext_vector_type(8)))  float    v8f;
typedef _Float16 h16;

#define DEV __device__ __forceinline__

DEV v8f wmma16(v16h a, v16h b, v8f c) {
  return __builtin_amdgcn_wmma_f32_16x16x32_f16(false, a, false, b,
                                                (short)0, c, false, false);
}

// Load a pre-swizzled fragment: p = this lane's 16 contiguous halfs (32B).
DEV v16h ld_frag(const h16* p) {
  union { uint4 q[2]; v16h v; } u;
  u.q[0] = *(const uint4*)p;
  u.q[1] = *(const uint4*)(p + 8);
  return u.v;
}

// A fragment 16(M)x32(K) f16 from row-major memory (16B-aligned rows).
DEV v16h load_a_f16(const h16* src, int lda) {
  int lane = threadIdx.x & 31;
  const h16* p = src + (long)(lane & 15) * lda + ((lane >> 4) << 3);
  union { uint4 q[2]; v16h v; } u;
  u.q[0] = *(const uint4*)p;
  u.q[1] = *(const uint4*)(p + 16);
  return u.v;
}

// C/D 16x16 f32: VGPR v holds row M = v + 8*(lane>=16), col = lane&15.
DEV void store_d_f16(h16* d, int ldd, v8f c) {
  int lane = threadIdx.x & 31;
  int col = lane & 15;
  int rb  = (lane >> 4) << 3;
#pragma unroll
  for (int v = 0; v < 8; ++v) d[(long)(v + rb) * ldd + col] = (h16)c[v];
}

// fragment-layout index helpers: (kk in [0,32), c in [0,16)) -> lane,elem
DEV int frag_lane(int kk, int c) { return c + (((kk >> 3) & 1) << 4); }
DEV int frag_elem(int kk)        { return (kk & 7) + ((kk >> 4) << 3); }

// ---------------- Tensor Data Mover: 1-D global -> LDS copy ----------------
// D# per CDNA5 ISA ch.8: group0 = count/lds_addr/global_addr/type,
// group1 = data_size + dims/strides, groups 2-3 unused (<=2D tensor).
typedef unsigned int u32x4 __attribute__((ext_vector_type(4)));
typedef int          i32x8 __attribute__((ext_vector_type(8)));
typedef int          i32x4 __attribute__((ext_vector_type(4)));

DEV void tdm_load_1d(unsigned int lds_byte_addr, const void* gptr,
                     int nelem_halfs) {
  unsigned long long ga = (unsigned long long)(uintptr_t)gptr;
  u32x4 g0;
  g0[0] = 1u;                                   // D# count = 1 (valid)
  g0[1] = lds_byte_addr;                        // LDS address (bytes)
  g0[2] = (unsigned int)(ga & 0xffffffffull);   // global_addr[31:0]
  g0[3] = (unsigned int)(ga >> 32) | (2u << 30);// global_addr[56:32] | type=2
  i32x8 g1;
  g1[0] = 1 << 16;                              // data_size = 2 bytes
  g1[1] = (int)((unsigned int)nelem_halfs << 16); // tensor_dim0[15:0]
  g1[2] = 1 << 16;                              // tensor_dim1 = 1
  g1[3] = (int)((unsigned int)nelem_halfs << 16); // tile_dim0
  g1[4] = 1;                                    // tile_dim1 = 1
  g1[5] = nelem_halfs;                          // tensor_dim0_stride
  g1[6] = 0;
  g1[7] = 0;
  i32x4 z4 = (i32x4)0;
#if defined(__clang_major__) && __clang_major__ >= 23
  i32x8 z8 = (i32x8)0;
  __builtin_amdgcn_tensor_load_to_lds(g0, g1, z4, z4, z8, 0);
#else
  __builtin_amdgcn_tensor_load_to_lds(g0, g1, z4, z4, 0);
#endif
}

// ---------------- basis offsets (halfs) ------------------------------------
static constexpr int A2R = 0;        // [32][128] fwd H cos
static constexpr int A2I = 4096;     // [32][128] fwd H -sin
static constexpr int A2IN = 8192;    // [32][128] fwd H +sin
static constexpr int A3R = 12288;    // [128][32] inv H cos
static constexpr int A3I = 16384;    // [128][32] inv H +sin
static constexpr int A3IN = 20480;   // [128][32] inv H -sin
static constexpr int BWFF = 24576;   // fwd W basis, frag layout [4kt][4nt][512]
static constexpr int BWIF = 32768;   // inv W basis, frag layout [2kt][8nt][512]

// ---------------- small kernels --------------------------------------------

__global__ void k_zero(float* gstats) { gstats[threadIdx.x] = 0.f; }

__global__ void k_emb(const float* t_emb, const float* cond,
                      const float* tp_w, const float* tp_b,
                      const float* cp_w, const float* cp_b,
                      float* tvec, float* scl, float* shf) {
  int b = blockIdx.x, t = threadIdx.x;
  __shared__ float st[512], sc[512];
  for (int i = t; i < 512; i += 256) {
    float a = t_emb[b * 512 + i]; st[i] = a / (1.f + expf(-a));
    float c = cond[b * 512 + i];  sc[i] = c / (1.f + expf(-c));
  }
  __syncthreads();
  float a0 = tp_b[t], a1 = cp_b[t], a2 = cp_b[t + 256];
  for (int k = 0; k < 512; ++k) {
    a0 += st[k] * tp_w[t * 512 + k];
    float s = sc[k];
    a1 += s * cp_w[t * 512 + k];
    a2 += s * cp_w[(t + 256) * 512 + k];
  }
  tvec[b * 256 + t] = a0; scl[b * 256 + t] = a1; shf[b * 256 + t] = a2;
}

__global__ void k_basis(h16* bs) {
  const float s  = 0.0883883476483184f;          // 1/sqrt(128) (ortho)
  const float tw = 6.283185307179586f / 128.f;
  for (int i = threadIdx.x; i < 40960; i += blockDim.x) {
    float v;
    if (i < 12288) {                             // A2R/A2I/A2IN [kh][h]
      int arr = i >> 12; int j = i & 4095;
      int kh = j >> 7, hh = j & 127;
      float th = tw * kh * hh;
      v = (arr == 0) ? cosf(th) * s : (arr == 1 ? -sinf(th) * s : sinf(th) * s);
    } else if (i < 24576) {                      // A3R/A3I/A3IN [h][kh]
      int j = i - 12288; int arr = j >> 12; j &= 4095;
      int hh = j >> 5, kh = j & 31;
      float th = tw * kh * hh;
      v = (arr == 0) ? cosf(th) * s : (arr == 1 ? sinf(th) * s : -sinf(th) * s);
    } else if (i < 32768) {                      // BWFF frag: row=w(128), col=n(64)
      int lc = i - 24576;
      int blk = lc >> 9, e = lc & 511;
      int kt = blk >> 2, nt = blk & 3;
      int L = e >> 4, j = e & 15;
      int k = kt * 32 + ((L >> 4) << 3) + (j & 7) + ((j >> 3) << 4); // w
      int n = nt * 16 + (L & 15);
      v = (n < 32) ? cosf(tw * n * k) * s : -sinf(tw * (n - 32) * k) * s;
    } else {                                     // BWIF frag: row=n(64), col=w(128)
      int lc = i - 32768;
      int blk = lc >> 9, e = lc & 511;
      int kt = blk >> 3, nt = blk & 7;
      int L = e >> 4, j = e & 15;
      int n = kt * 32 + ((L >> 4) << 3) + (j & 7) + ((j >> 3) << 4);
      int w = nt * 16 + (L & 15);
      int k0 = (n < 32) ? n : n - 32;
      float cn = (k0 == 0) ? 1.f : 2.f;
      v = (n < 32) ? cn * cosf(tw * k0 * w) * s : -cn * sinf(tw * k0 * w) * s;
    }
    bs[i] = (h16)v;
  }
}

__global__ void k_cvt(const float* x, h16* xh) {
  long i4 = (long)blockIdx.x * blockDim.x + threadIdx.x;
  float4 v = ((const float4*)x)[i4];
  h16* d = xh + i4 * 4;
  d[0] = (h16)v.x; d[1] = (h16)v.y; d[2] = (h16)v.z; d[3] = (h16)v.w;
}

// pw_w f32 [o][i] -> f16 A-fragment layout [mt(16)][kt(8)][lane(32)][16]
__global__ void k_pwa(const float* pw_w, h16* pwA) {
  for (int idx = blockIdx.x * 256 + threadIdx.x; idx < 65536;
       idx += gridDim.x * 256) {
    int j = idx & 15, L = (idx >> 4) & 31, kt = (idx >> 9) & 7, mt = idx >> 12;
    int row = L & 15;
    int k = ((L >> 4) << 3) + (j & 7) + ((j >> 3) << 4);
    pwA[idx] = (h16)pw_w[(mt * 16 + row) * 256 + kt * 32 + k];
  }
}

// spectral weights [C*C][1024] f32 -> per-mode B-fragment layout f16
__global__ void k_wt(const float* wr, const float* wi, h16* WrT, h16* WiT) {
  const float* src = blockIdx.z ? wi : wr;
  h16* dst = blockIdx.z ? WiT : WrT;
  __shared__ float tile[32][33];
  int m0 = blockIdx.x * 32, r0 = blockIdx.y * 32;  // r0: 32 consecutive cc
  for (int j = threadIdx.y; j < 32; j += 8)
    tile[j][threadIdx.x] = src[(long)(r0 + j) * 1024 + m0 + threadIdx.x];
  __syncthreads();
  int i_ch = r0 >> 8;                              // constant in-channel
  int o0 = r0 & 255;
  int kt = i_ch >> 5, kk = i_ch & 31;
  int Lb = ((kk >> 3) & 1) << 4;
  int jj = frag_elem(kk);
  for (int j = threadIdx.y; j < 32; j += 8) {
    int o = o0 + threadIdx.x;
    int nt = o >> 4, L = (o & 15) + Lb;
    dst[(long)(m0 + j) * 65536 + (kt * 16 + nt) * 512 + L * 16 + jj] =
        (h16)tile[threadIdx.x][j];
  }
}

// ---------------- forward DFT along W --------------------------------------
__global__ void k_fwdw(const h16* xh, const h16* bwff, h16* Y) {
  int img = blockIdx.x;
  int mt = threadIdx.x >> 5;                       // 8 waves -> 8 h-tiles
  int lane = threadIdx.x & 31;
  long row0 = (long)img * 128 + mt * 16;
  v8f acc[4] = {};
#pragma unroll
  for (int kt = 0; kt < 4; ++kt) {
    v16h a = load_a_f16(xh + row0 * 128 + kt * 32, 128);
#pragma unroll
    for (int nt = 0; nt < 4; ++nt) {
      v16h b = ld_frag(bwff + (kt * 4 + nt) * 512 + lane * 16);
      acc[nt] = wmma16(a, b, acc[nt]);
    }
  }
  h16* Yb = Y + (long)img * 8192;
  int col = lane & 15, rb = (lane >> 4) << 3;
  int kt2 = mt >> 1, ksub = mt & 1;
#pragma unroll
  for (int nt = 0; nt < 4; ++nt)
#pragma unroll
    for (int v = 0; v < 8; ++v) {
      int kk = ksub * 16 + v + rb;
      Yb[(kt2 * 4 + nt) * 512 + frag_lane(kk, col) * 16 + frag_elem(kk)] =
          (h16)acc[nt][v];
    }
}

// ---------------- forward DFT along H (complex) ----------------------------
__global__ void k_fwdh(const h16* Y, const h16* a2r, const h16* a2i,
                       const h16* a2in, h16* Ztr, h16* Zti, h16* Ztin) {
  int img = blockIdx.x;
  int wave = threadIdx.x >> 5;
  int lane = threadIdx.x & 31;
  int mt = wave & 1, pair = wave >> 1;             // pair0: Zr, pair1: Zi
  const h16* A1 = a2r + mt * 16 * 128;
  const h16* A2 = (pair ? a2i : a2in) + mt * 16 * 128;
  const h16* Yb = Y + (long)img * 8192;
  v8f acc[2] = {};
#pragma unroll
  for (int kt = 0; kt < 4; ++kt) {
    v16h va = load_a_f16(A1 + kt * 32, 128);
    v16h vb = load_a_f16(A2 + kt * 32, 128);
#pragma unroll
    for (int nt = 0; nt < 2; ++nt) {
      int ct1 = pair ? 2 + nt : nt;                // Yr cols 0..1, Yi cols 2..3
      int ct2 = pair ? nt : 2 + nt;
      v16h f1 = ld_frag(Yb + (kt * 4 + ct1) * 512 + lane * 16);
      v16h f2 = ld_frag(Yb + (kt * 4 + ct2) * 512 + lane * 16);
      acc[nt] = wmma16(va, f1, acc[nt]);
      acc[nt] = wmma16(vb, f2, acc[nt]);
    }
  }
  int col = lane & 15, rb = (lane >> 4) << 3;
  int bb = img >> 8, cc = img & 255;
#pragma unroll
  for (int nt = 0; nt < 2; ++nt)
#pragma unroll
    for (int v = 0; v < 8; ++v) {
      int kh = mt * 16 + v + rb, ky = nt * 16 + col;
      long ad = (((long)(kh * 32 + ky) * 16 + bb) << 8) + cc;
      float val = acc[nt][v];
      if (pair == 0) Ztr[ad] = (h16)val;
      else { Zti[ad] = (h16)val; Ztin[ad] = (h16)(-val); }
    }
}

// ---------------- per-mode spectral channel mix ----------------------------
// A matrices (Zr/Zi/-Zi, 8KB each) staged into LDS via the Tensor Data Mover;
// weight stream (268MB) consumed directly with b128 frag loads + prefetch.
__global__ void k_spec(const h16* Ztr, const h16* Zti, const h16* Ztin,
                       const h16* WrT, const h16* WiT, h16* OtR, h16* OtI) {
  __shared__ __align__(16) h16 sZ[3 * 4096];
  int mode = blockIdx.x;
  int wave = threadIdx.x >> 5;
  int lane = threadIdx.x & 31;
  if (wave == 0) {                                 // one wave issues 3 DMAs
    unsigned int lds0 = (unsigned int)(uintptr_t)(void*)&sZ[0];
    tdm_load_1d(lds0,         Ztr + (long)mode * 4096, 4096);
    tdm_load_1d(lds0 + 8192,  Zti + (long)mode * 4096, 4096);
    tdm_load_1d(lds0 + 16384, Ztin + (long)mode * 4096, 4096);
    __builtin_amdgcn_s_wait_tensorcnt(0);
  }
  __syncthreads();
  const h16* Ar = sZ;
  const h16* Ai = sZ + 4096;
  const h16* An = sZ + 8192;
  const h16* Br = WrT + (long)mode * 65536;
  const h16* Bi = WiT + (long)mode * 65536;
  v8f orr[2] = {}, oii[2] = {};
#pragma unroll
  for (int kt = 0; kt < 8; ++kt) {
    if (kt < 7) {                                  // stream next weight block
      __builtin_prefetch(Br + (long)(kt + 1) * 16 * 512 + wave * 2 * 512, 0, 0);
      __builtin_prefetch(Bi + (long)(kt + 1) * 16 * 512 + wave * 2 * 512, 0, 0);
    }
    v16h azr = load_a_f16(Ar + kt * 32, 256);
    v16h azi = load_a_f16(Ai + kt * 32, 256);
    v16h azn = load_a_f16(An + kt * 32, 256);
#pragma unroll
    for (int q = 0; q < 2; ++q) {
      int nt = wave * 2 + q;
      v16h br = ld_frag(Br + (kt * 16 + nt) * 512 + lane * 16);
      v16h bi = ld_frag(Bi + (kt * 16 + nt) * 512 + lane * 16);
      orr[q] = wmma16(azr, br, orr[q]);
      orr[q] = wmma16(azn, bi, orr[q]);            // -Zi * Wi
      oii[q] = wmma16(azr, bi, oii[q]);
      oii[q] = wmma16(azi, br, oii[q]);
    }
  }
#pragma unroll
  for (int q = 0; q < 2; ++q) {
    int nt = wave * 2 + q;
    store_d_f16(OtR + (long)mode * 4096 + nt * 16, 256, orr[q]);
    store_d_f16(OtI + (long)mode * 4096 + nt * 16, 256, oii[q]);
  }
}

// mode-plane transpose [1024][4096] -> per-image B fragments
__global__ void k_ot(const h16* OtR, const h16* OtI, h16* Or, h16* Oi) {
  const h16* src = blockIdx.z ? OtI : OtR;
  h16* dst = blockIdx.z ? Oi : Or;
  __shared__ h16 tile[32][33];
  int c0 = blockIdx.x * 32;                        // image (bo) block
  int kh = blockIdx.y;                             // 32 modes = one kh row
  int r0 = kh * 32;
  for (int j = threadIdx.y; j < 32; j += 8)
    tile[j][threadIdx.x] = src[(long)(r0 + j) * 4096 + c0 + threadIdx.x];
  __syncthreads();
  int Lb = ((kh >> 3) & 1) << 4;
  int jj = frag_elem(kh);
  for (int j = threadIdx.y; j < 32; j += 8) {
    int ky = threadIdx.x;
    int nt = ky >> 4, L = (ky & 15) + Lb;
    dst[(long)(c0 + j) * 1024 + nt * 512 + L * 16 + jj] = tile[threadIdx.x][j];
  }
}

// ---------------- inverse DFT along H (complex, K=32) ----------------------
__global__ void k_invh(const h16* Or, const h16* Oi, const h16* a3r,
                       const h16* a3i, const h16* a3in, h16* T) {
  int img = blockIdx.x;
  int mt = threadIdx.x >> 5;                       // 8 waves -> 8 h-tiles
  int lane = threadIdx.x & 31;
  v16h ar = load_a_f16(a3r + mt * 16 * 32, 32);
  v16h ai = load_a_f16(a3i + mt * 16 * 32, 32);
  v16h an = load_a_f16(a3in + mt * 16 * 32, 32);
  const h16* Obr = Or + (long)img * 1024;
  const h16* Obi = Oi + (long)img * 1024;
  v8f tr[2] = {}, ti[2] = {};
#pragma unroll
  for (int nt = 0; nt < 2; ++nt) {
    v16h br = ld_frag(Obr + nt * 512 + lane * 16);
    v16h bi = ld_frag(Obi + nt * 512 + lane * 16);
    tr[nt] = wmma16(ar, br, tr[nt]);
    tr[nt] = wmma16(an, bi, tr[nt]);               // -Gi*Oi
    ti[nt] = wmma16(ar, bi, ti[nt]);
    ti[nt] = wmma16(ai, br, ti[nt]);
  }
  h16* Tb = T + (long)img * 128 * 64 + mt * 16 * 64;
#pragma unroll
  for (int nt = 0; nt < 2; ++nt) {
    store_d_f16(Tb + nt * 16, 64, tr[nt]);
    store_d_f16(Tb + 32 + nt * 16, 64, ti[nt]);
  }
}

// ---------------- inverse real DFT along W (K=64, N=128) -------------------
__global__ void k_invw(const h16* T, const h16* bwif, h16* S) {
  int wave = threadIdx.x >> 5;
  int lane = threadIdx.x & 31;
  long row0 = (long)blockIdx.x * 128 + wave * 16;
  v8f acc[8] = {};
#pragma unroll
  for (int kt = 0; kt < 2; ++kt) {
    v16h a = load_a_f16(T + row0 * 64 + kt * 32, 64);
#pragma unroll
    for (int nt = 0; nt < 8; ++nt) {
      v16h b = ld_frag(bwif + (kt * 8 + nt) * 512 + lane * 16);
      acc[nt] = wmma16(a, b, acc[nt]);
    }
  }
#pragma unroll
  for (int nt = 0; nt < 8; ++nt)
    store_d_f16(S + row0 * 128 + nt * 16, 128, acc[nt]);
}

// ---------------- pointwise conv GEMM + fused epilogue + GN partials -------
__global__ void k_pw(const h16* xh, const h16* pwA, const float* pw_b,
                     const h16* S, const float* tvec, const float* scl,
                     const float* shf, float* hbuf, float* gstats) {
  // activation tile staged in LDS already frag-swizzled; lane stride 24 halfs
  __shared__ __align__(16) h16 sB[4 * 32 * 24];
  __shared__ float sgs[8][2];
  int b = blockIdx.x, s0 = blockIdx.y * 64;
  int tid = threadIdx.x, wave = tid >> 5, lane = tid & 31;
  if (tid < 16) ((float*)sgs)[tid] = 0.f;
  v8f acc[2][4] = {};
  for (int kt = 0; kt < 8; ++kt) {
    __syncthreads();
    {   // stage 32(K) x 64(N) f16 tile, swizzling into fragment layout
      int base = tid * 8, k = base >> 6, n = base & 63;
      const h16* srcp = xh + ((long)(b * 256 + kt * 32 + k) << 14) + s0 + n;
      if (kt < 7)
        __builtin_prefetch(srcp + (32l << 14), 0, 0);
      int nt = n >> 4, Lb = ((k >> 3) & 1) << 4, j = frag_elem(k);
      int c0 = n & 15;
#pragma unroll
      for (int e = 0; e < 8; ++e)
        sB[nt * 768 + (c0 + e + Lb) * 24 + j] = srcp[e];
    }
    __syncthreads();
    v16h a0 = ld_frag(pwA + ((long)wave * 8 + kt) * 512 + lane * 16);
    v16h a1 = ld_frag(pwA + ((long)(wave + 8) * 8 + kt) * 512 + lane * 16);
#pragma unroll
    for (int nt = 0; nt < 4; ++nt) {
      v16h bf = ld_frag(sB + nt * 768 + lane * 24);
      acc[0][nt] = wmma16(a0, bf, acc[0][nt]);
      acc[1][nt] = wmma16(a1, bf, acc[1][nt]);
    }
  }
  int col = lane & 15, rb = (lane >> 4) << 3;
#pragma unroll
  for (int mi = 0; mi < 2; ++mi) {
    int mtb = (mi ? wave + 8 : wave) * 16;
    int g = mtb >> 5;
#pragma unroll
    for (int nt = 0; nt < 4; ++nt) {
      float ls = 0.f, lq = 0.f;
#pragma unroll
      for (int v = 0; v < 8; ++v) {
        int o = mtb + v + rb, bo = (b << 8) + o;
        long idx = ((long)bo << 14) + s0 + nt * 16 + col;
        float hv = acc[mi][nt][v] + (float)S[idx] + pw_b[o] + tvec[bo];
        hv = hv * (1.f + scl[bo]) + shf[bo];
        hbuf[idx] = hv;
        ls += hv; lq += hv * hv;
      }
      atomicAdd(&sgs[g][0], ls);                   // ds_add_f32
      atomicAdd(&sgs[g][1], lq);
    }
  }
  __syncthreads();
  if (tid < 8) {
    atomicAdd(&gstats[(b * 8 + tid) * 2 + 0], sgs[tid][0]);
    atomicAdd(&gstats[(b * 8 + tid) * 2 + 1], sgs[tid][1]);
  }
}

// ---------------- groupnorm finalize + GELU + residual ---------------------
__global__ void k_gn(const float* hbuf, const float* x, const float* gstats,
                     const float* gg, const float* gb, float* out) {
  long i4 = (long)blockIdx.x * blockDim.x + threadIdx.x;
  long idx = i4 * 4;
  int b = (int)(idx >> 22), o = (int)((idx >> 14) & 255);
  int bg = b * 8 + (o >> 5);
  const float n = 524288.f;                        // 32*128*128 per group
  float mu = gstats[bg * 2] / n;
  float var = gstats[bg * 2 + 1] / n - mu * mu;
  float rs = rsqrtf(var + 1e-5f);
  float ga = gg[o], be = gb[o];
  float4 hv = ((const float4*)hbuf)[i4];
  float4 xv = ((const float4*)x)[i4];
  float4 ov;
  const float is2 = 0.7071067811865476f;
  float hn;
  hn = (hv.x - mu) * rs * ga + be; ov.x = xv.x + 0.5f * hn * (1.f + erff(hn * is2));
  hn = (hv.y - mu) * rs * ga + be; ov.y = xv.y + 0.5f * hn * (1.f + erff(hn * is2));
  hn = (hv.z - mu) * rs * ga + be; ov.z = xv.z + 0.5f * hn * (1.f + erff(hn * is2));
  hn = (hv.w - mu) * rs * ga + be; ov.w = xv.w + 0.5f * hn * (1.f + erff(hn * is2));
  ((float4*)out)[i4] = ov;
}

// ---------------------------------------------------------------------------
extern "C" void kernel_launch(void* const* d_in, const int* in_sizes, int n_in,
                              void* d_out, int out_size, void* d_ws, size_t ws_size,
                              hipStream_t stream) {
  const float* x      = (const float*)d_in[0];
  const float* t_emb  = (const float*)d_in[1];
  const float* cond   = (const float*)d_in[2];
  const float* w_real = (const float*)d_in[3];
  const float* w_imag = (const float*)d_in[4];
  const float* pw_w   = (const float*)d_in[5];
  const float* pw_b   = (const float*)d_in[6];
  const float* tp_w   = (const float*)d_in[7];
  const float* tp_b   = (const float*)d_in[8];
  const float* cp_w   = (const float*)d_in[9];
  const float* cp_b   = (const float*)d_in[10];
  const float* gn_g   = (const float*)d_in[11];
  const float* gn_b   = (const float*)d_in[12];
  float* out = (float*)d_out;

  char* w = (char*)d_ws;
  size_t off = 0;
  auto alloc = [&](size_t bytes) -> void* {
    void* p = w + off;
    off = (off + bytes + 255) & ~(size_t)255;
    return p;
  };
  h16* xh   = (h16*)alloc(134217728);   // x as f16
  h16* WrT  = (h16*)alloc(134217728);   // [1024][frag 65536] f16
  h16* WiT  = (h16*)alloc(134217728);
  h16* Y    = (h16*)alloc(67108864);    // per-image frag layout
  h16* Ztr  = (h16*)alloc(8388608);     // [mode][b][i]
  h16* Zti  = (h16*)alloc(8388608);
  h16* Ztin = (h16*)alloc(8388608);
  h16* OtR  = (h16*)alloc(8388608);     // [mode][b*256+o]
  h16* OtI  = (h16*)alloc(8388608);
  h16* Or   = (h16*)alloc(8388608);     // per-image frag layout
  h16* Oi   = (h16*)alloc(8388608);
  h16* T    = (h16*)alloc(67108864);    // [b,c,h][64]
  h16* S    = (h16*)alloc(134217728);   // irfft2 result, f16
  float* hbuf   = (float*)alloc(268435456);
  float* tvec   = (float*)alloc(16384);
  float* scl    = (float*)alloc(16384);
  float* shf    = (float*)alloc(16384);
  h16*   bs     = (h16*)alloc(81920);
  h16*   pwA    = (h16*)alloc(131072);  // pw_w A-fragment layout
  float* gstats = (float*)alloc(1024);

  k_zero<<<1, 256, 0, stream>>>(gstats);
  k_emb<<<16, 256, 0, stream>>>(t_emb, cond, tp_w, tp_b, cp_w, cp_b,
                                tvec, scl, shf);
  k_basis<<<1, 256, 0, stream>>>(bs);
  k_cvt<<<65536, 256, 0, stream>>>(x, xh);
  k_pwa<<<64, 256, 0, stream>>>(pw_w, pwA);
  k_wt<<<dim3(32, 2048, 2), dim3(32, 8), 0, stream>>>(w_real, w_imag, WrT, WiT);

  k_fwdw<<<4096, 256, 0, stream>>>(xh, bs + BWFF, Y);
  k_fwdh<<<4096, 128, 0, stream>>>(Y, bs + A2R, bs + A2I, bs + A2IN,
                                   Ztr, Zti, Ztin);
  k_spec<<<1024, 256, 0, stream>>>(Ztr, Zti, Ztin, WrT, WiT, OtR, OtI);
  k_ot<<<dim3(128, 32, 2), dim3(32, 8), 0, stream>>>(OtR, OtI, Or, Oi);
  k_invh<<<4096, 256, 0, stream>>>(Or, Oi, bs + A3R, bs + A3I, bs + A3IN, T);
  k_invw<<<4096, 256, 0, stream>>>(T, bs + BWIF, S);
  k_pw<<<dim3(16, 256), 256, 0, stream>>>(xh, pwA, pw_b, S, tvec, scl, shf,
                                          hbuf, gstats);
  k_gn<<<65536, 256, 0, stream>>>(hbuf, x, gstats, gn_g, gn_b, out);
}
